// LowRankSVDBlock_6786048327901
// MI455X (gfx1250) — matrix operand order repaired and checked by reading, not compiled, not run on hardware
//
#include <hip/hip_runtime.h>

typedef __attribute__((ext_vector_type(16))) _Float16 v16h;
typedef __attribute__((ext_vector_type(8)))  float    v8f;

#if __has_builtin(__builtin_amdgcn_global_load_async_to_lds_b128)
#define HAVE_ASYNC_LDS 1
typedef int v4i_ __attribute__((vector_size(16)));
typedef __attribute__((address_space(1))) v4i_* gv4i_p;
typedef __attribute__((address_space(3))) v4i_* lv4i_p;
#else
#define HAVE_ASYNC_LDS 0
#endif

static __device__ __forceinline__ v8f wmma_f16(v16h a, v16h b, v8f c) {
    return __builtin_amdgcn_wmma_f32_16x16x32_f16(
        false, a, false, b, (short)0, c, false, false);
}

// 16 contiguous halves (32B) -> one fragment register block
static __device__ __forceinline__ v16h load_frag16(const _Float16* __restrict__ p) {
    v16h v;
    #pragma unroll
    for (int i = 0; i < 16; ++i) v[i] = p[i];
    return v;
}
// A-fragment: rowp points at row start + k0; kb = lane-dependent sub-base
static __device__ __forceinline__ v16h load_a_frag(const _Float16* __restrict__ rowp, int kb) {
    v16h a;
    #pragma unroll
    for (int i = 0; i < 8; ++i) a[i]     = rowp[kb + i];
    #pragma unroll
    for (int i = 0; i < 8; ++i) a[8 + i] = rowp[8 + kb + i];
    return a;
}

// ---------------------------------------------------------------------------
// Weight packing kernels (run once per call; weights ~20MB, L2-resident)
// ---------------------------------------------------------------------------
// in [K][N] f32 -> out [N][K] f16   (transpose+convert: B-fragment friendly)
__global__ void __launch_bounds__(256)
pack_transpose_f16(const float* __restrict__ in, _Float16* __restrict__ out,
                   int K, int N) {
    long idx = (long)blockIdx.x * blockDim.x + threadIdx.x;
    if (idx >= (long)K * N) return;
    int n = (int)(idx / K), k = (int)(idx % K);
    out[idx] = (_Float16)in[(long)k * N + n];
}

// U [D][H*r] f32 -> out [H*64][D] f16, rows h*64+rr, zero-padded rr>=r
__global__ void __launch_bounds__(256)
pack_U_pad(const float* __restrict__ U, _Float16* __restrict__ out,
           int D, int H, int r) {
    long idx = (long)blockIdx.x * blockDim.x + threadIdx.x;
    if (idx >= (long)H * 64 * D) return;
    int n = (int)(idx / D), d = (int)(idx % D);
    int h = n >> 6, rr = n & 63;
    out[idx] = (rr < r) ? (_Float16)U[(long)d * (H * r) + h * r + rr] : (_Float16)0.f;
}

// W [H][r][64] f32 -> out [H][64][64] f16 (d-major, k-minor), zero k>=r
__global__ void __launch_bounds__(256)
pack_headW_pad(const float* __restrict__ W, _Float16* __restrict__ out,
               int H, int r) {
    long idx = (long)blockIdx.x * blockDim.x + threadIdx.x;
    if (idx >= (long)H * 64 * 64) return;
    int h = (int)(idx >> 12), d = (int)((idx >> 6) & 63), kk = (int)(idx & 63);
    out[idx] = (kk < r) ? (_Float16)W[((long)h * r + kk) * 64 + d] : (_Float16)0.f;
}

// ---------------------------------------------------------------------------
// LayerNorm (f32 in) -> f16 out.  One block per row, D=1024.
// ---------------------------------------------------------------------------
__global__ void __launch_bounds__(256)
ln_to_f16(const float* __restrict__ x, const float* __restrict__ g,
          const float* __restrict__ b, _Float16* __restrict__ out, int D) {
    const int row = blockIdx.x;
    const float* xr = x + (long)row * D;
    float s = 0.f, s2 = 0.f;
    for (int i = threadIdx.x; i < D; i += blockDim.x) {
        float v = xr[i]; s += v; s2 += v * v;
    }
    __shared__ float red0[8], red1[8];
    #pragma unroll
    for (int off = 16; off > 0; off >>= 1) {
        s += __shfl_xor(s, off); s2 += __shfl_xor(s2, off);
    }
    int wid = threadIdx.x >> 5, lid = threadIdx.x & 31;
    if (lid == 0) { red0[wid] = s; red1[wid] = s2; }
    __syncthreads();
    if (wid == 0) {
        s  = (lid < 8) ? red0[lid] : 0.f;
        s2 = (lid < 8) ? red1[lid] : 0.f;
        #pragma unroll
        for (int off = 4; off > 0; off >>= 1) {
            s += __shfl_xor(s, off); s2 += __shfl_xor(s2, off);
        }
        if (lid == 0) { red0[0] = s; red1[0] = s2; }
    }
    __syncthreads();
    float mean = red0[0] / (float)D;
    float rstd = rsqrtf(red1[0] / (float)D - mean * mean + 1e-5f);
    _Float16* outr = out + (long)row * D;
    for (int i = threadIdx.x; i < D; i += blockDim.x)
        outr[i] = (_Float16)((xr[i] - mean) * rstd * g[i] + b[i]);
}

// ---------------------------------------------------------------------------
// WMMA GEMM: C[M,N] = A(f16,[M,K]) @ Bt(f16,[N,K])^T  (+bias)(+gelu)(+res)
// Block = 128 threads (4 waves); block tile 128(M) x 64(N); wave tile 32x64.
// B tile (64x32 f16, 4KB) staged to LDS via GLOBAL_LOAD_ASYNC_TO_LDS_B128.
// ---------------------------------------------------------------------------
template<bool BIAS, bool GELU, bool RES, bool OUT16>
__global__ void __launch_bounds__(128)
gemm_wmma(const _Float16* __restrict__ A, const _Float16* __restrict__ Bt,
          const float* __restrict__ bias, const float* __restrict__ res,
          _Float16* __restrict__ out16, float* __restrict__ out32,
          int M, int N, int K) {
    __shared__ _Float16 Blds[64][32];
    const int nblk = blockIdx.x * 64;
    const int mblk = blockIdx.y * 128;
    const int lane = threadIdx.x & 31;
    const int wave = threadIdx.x >> 5;
    const int m0   = mblk + wave * 32;

    const int am    = lane & 15;
    const int kb    = (lane < 16) ? 0 : 16;
    const int koff  = (lane < 16) ? 0 : 16;
    const int nE    = lane & 15;
    const int mbase = (lane < 16) ? 0 : 8;

    // staging: thread t copies one 16B chunk: LDS row sn, half-offset sp
    const int sn = threadIdx.x >> 1;
    const int sp = (threadIdx.x & 1) * 8;

    v8f acc0[4] = {}, acc1[4] = {};
    const _Float16* arow0 = A + (long)(m0 + am) * K;
    const _Float16* arow1 = A + (long)(m0 + 16 + am) * K;
    const _Float16* brow  = Bt + (long)(nblk + sn) * K + sp;

    for (int k0 = 0; k0 < K; k0 += 32) {
        __syncthreads();
#if HAVE_ASYNC_LDS
        __builtin_amdgcn_global_load_async_to_lds_b128(
            (gv4i_p)(brow + k0), (lv4i_p)&Blds[sn][sp], 0, 0);
#if __has_builtin(__builtin_amdgcn_s_wait_asynccnt)
        __builtin_amdgcn_s_wait_asynccnt(0);
#else
        asm volatile("s_wait_asynccnt 0" ::: "memory");
#endif
#else
        *(uint4*)&Blds[sn][sp] = *(const uint4*)(brow + k0);
#endif
        __syncthreads();

        __builtin_prefetch(arow0 + k0 + 32, 0, 3);
        __builtin_prefetch(arow1 + k0 + 32, 0, 3);
        v16h a0 = load_a_frag(arow0 + k0, kb);
        v16h a1 = load_a_frag(arow1 + k0, kb);
        #pragma unroll
        for (int f = 0; f < 4; ++f) {
            v16h bf = load_frag16(&Blds[f * 16 + nE][koff]);
            acc0[f] = wmma_f16(a0, bf, acc0[f]);
            acc1[f] = wmma_f16(a1, bf, acc1[f]);
        }
    }

    #pragma unroll
    for (int sub = 0; sub < 2; ++sub) {
        #pragma unroll
        for (int f = 0; f < 4; ++f) {
            #pragma unroll
            for (int i = 0; i < 8; ++i) {
                int r = m0 + sub * 16 + mbase + i;
                int c = nblk + f * 16 + nE;
                float v = sub ? acc1[f][i] : acc0[f][i];
                if (BIAS) v += bias[c];
                if (GELU) {
                    float u = v;
                    v = 0.5f * u * (1.0f + tanhf(0.7978845608f * (u + 0.044715f * u * u * u)));
                }
                if (RES) v += res[(long)r * N + c];
                if (OUT16) out16[(long)r * N + c] = (_Float16)v;
                else       out32[(long)r * N + c] = v;
            }
        }
    }
}

// ---------------------------------------------------------------------------
// Per-head reconstruction from padded latents.
// A = Ppad [B*S, 3072] (per-head K=64 slice at aoff+h*64), W = [H][64][64] f16.
// Wave tile 32(M) x 64(dh).  TRANS: out [BH,64,S] (for V), else [BH,S,64].
// ---------------------------------------------------------------------------
template<bool TRANS>
__global__ void __launch_bounds__(128)
head_proj(const _Float16* __restrict__ P, int ldp, int aoff,
          const _Float16* __restrict__ W, const float* __restrict__ bias,
          _Float16* __restrict__ out, int S, int H) {
    const int h    = blockIdx.y;
    const int lane = threadIdx.x & 31;
    const int wave = threadIdx.x >> 5;
    const int m0   = blockIdx.x * 128 + wave * 32;

    const int am    = lane & 15;
    const int kb    = (lane < 16) ? 0 : 16;
    const int koff  = (lane < 16) ? 0 : 16;
    const int nE    = lane & 15;
    const int mbase = (lane < 16) ? 0 : 8;

    v8f acc0[4] = {}, acc1[4] = {};
    const _Float16* arow0 = P + (long)(m0 + am) * ldp + aoff + h * 64;
    const _Float16* arow1 = P + (long)(m0 + 16 + am) * ldp + aoff + h * 64;
    const _Float16* Wh    = W + (long)h * 64 * 64;

    #pragma unroll
    for (int k0 = 0; k0 < 64; k0 += 32) {
        v16h a0 = load_a_frag(arow0 + k0, kb);
        v16h a1 = load_a_frag(arow1 + k0, kb);
        #pragma unroll
        for (int f = 0; f < 4; ++f) {
            v16h bf = load_frag16(Wh + (long)(f * 16 + nE) * 64 + k0 + koff);
            acc0[f] = wmma_f16(a0, bf, acc0[f]);
            acc1[f] = wmma_f16(a1, bf, acc1[f]);
        }
    }

    const int b_ = m0 / S;
    const long base = (long)(b_ * H + h);
    #pragma unroll
    for (int sub = 0; sub < 2; ++sub) {
        #pragma unroll
        for (int f = 0; f < 4; ++f) {
            #pragma unroll
            for (int i = 0; i < 8; ++i) {
                int s = (m0 % S) + sub * 16 + mbase + i;
                int d = f * 16 + nE;
                float v = (sub ? acc1[f][i] : acc0[f][i]) + bias[h * 64 + d];
                if (TRANS) out[(base * 64 + d) * (long)S + s] = (_Float16)v;
                else       out[(base * (long)S + s) * 64 + d] = (_Float16)v;
            }
        }
    }
}

// ---------------------------------------------------------------------------
// Flash attention (causal), dh=64.  Q,K: [BH,S,64], Vt: [BH,64,S] (f16).
// Each wave: 16 query rows, key tiles of 32, online softmax via shuffles.
// ---------------------------------------------------------------------------
__global__ void __launch_bounds__(128)
flash_attn(const _Float16* __restrict__ Q, const _Float16* __restrict__ Kc,
           const _Float16* __restrict__ Vt, _Float16* __restrict__ ctx,
           int S, int H) {
    __shared__ _Float16 Plds[4][16][32];
    const int bh   = blockIdx.y;
    const int lane = threadIdx.x & 31;
    const int wave = threadIdx.x >> 5;
    const int q0   = blockIdx.x * 64 + wave * 16;

    const _Float16* Qp = Q  + ((long)bh * S + q0) * 64;
    const _Float16* Kp = Kc + (long)bh * S * 64;
    const _Float16* Vp = Vt + (long)bh * 64 * S;

    const int am    = lane & 15;
    const int kb    = (lane < 16) ? 0 : 16;
    const int koff  = (lane < 16) ? 0 : 16;
    const int nE    = lane & 15;
    const int mbase = (lane < 16) ? 0 : 8;

    v16h qa[2];
    #pragma unroll
    for (int c = 0; c < 2; ++c) qa[c] = load_a_frag(Qp + am * 64 + c * 32, kb);

    v8f acc[4] = {};
    float mrow[8], lrow[8];
    #pragma unroll
    for (int i = 0; i < 8; ++i) { mrow[i] = -1e30f; lrow[i] = 0.f; }
    const float scale = 0.125f;
    const int qmax = q0 + 15;

    for (int t0 = 0; t0 <= qmax; t0 += 32) {
        v8f s0 = {}, s1 = {};
        #pragma unroll
        for (int c = 0; c < 2; ++c) {
            v16h b0 = load_frag16(Kp + (long)(t0 + nE) * 64      + c * 32 + koff);
            v16h b1 = load_frag16(Kp + (long)(t0 + 16 + nE) * 64 + c * 32 + koff);
            s0 = wmma_f16(qa[c], b0, s0);
            s1 = wmma_f16(qa[c], b1, s1);
        }
        #pragma unroll
        for (int i = 0; i < 8; ++i) {
            int q = q0 + mbase + i;
            float v0 = (t0 + nE      <= q) ? s0[i] * scale : -1e30f;
            float v1 = (t0 + 16 + nE <= q) ? s1[i] * scale : -1e30f;
            float mx = fmaxf(v0, v1);
            mx = fmaxf(mx, __shfl_xor(mx, 1));
            mx = fmaxf(mx, __shfl_xor(mx, 2));
            mx = fmaxf(mx, __shfl_xor(mx, 4));
            mx = fmaxf(mx, __shfl_xor(mx, 8));
            float mN = fmaxf(mrow[i], mx);
            float p0 = __expf(v0 - mN);
            float p1 = __expf(v1 - mN);
            float ps = p0 + p1;
            ps += __shfl_xor(ps, 1);
            ps += __shfl_xor(ps, 2);
            ps += __shfl_xor(ps, 4);
            ps += __shfl_xor(ps, 8);
            float alpha = __expf(mrow[i] - mN);
            lrow[i] = lrow[i] * alpha + ps;
            mrow[i] = mN;
            #pragma unroll
            for (int f = 0; f < 4; ++f) acc[f][i] *= alpha;
            Plds[wave][mbase + i][nE]      = (_Float16)p0;
            Plds[wave][mbase + i][16 + nE] = (_Float16)p1;
        }
        v16h pa = load_a_frag(&Plds[wave][am][0], kb);
        #pragma unroll
        for (int f = 0; f < 4; ++f) {
            v16h vb = load_frag16(Vp + (long)(f * 16 + nE) * S + t0 + koff);
            acc[f] = wmma_f16(pa, vb, acc[f]);
        }
    }

    const int b_ = bh / H, h = bh % H;
    #pragma unroll
    for (int f = 0; f < 4; ++f) {
        #pragma unroll
        for (int i = 0; i < 8; ++i) {
            int s = q0 + mbase + i;
            float v = acc[f][i] / lrow[i];
            ctx[((long)(b_ * S + s)) * (64 * H) + h * 64 + f * 16 + nE] = (_Float16)v;
        }
    }
}

// ---------------------------------------------------------------------------
// Host-side orchestration
// ---------------------------------------------------------------------------
extern "C" void kernel_launch(void* const* d_in, const int* in_sizes, int n_in,
                              void* d_out, int out_size, void* d_ws, size_t ws_size,
                              hipStream_t stream) {
    (void)in_sizes; (void)n_in; (void)out_size; (void)ws_size;
    constexpr int B = 2, S = 2048, D = 1024, H = 16, dh = 64;
    constexpr int r = 48, r_out = 768, I = 4096, r_mlp = 512;
    constexpr int BS = B * S, BH = B * H;

    const float* hidden = (const float*)d_in[0];
    const float* ln1_g = (const float*)d_in[1];
    const float* ln1_b = (const float*)d_in[2];
    const float* ln2_g = (const float*)d_in[3];
    const float* ln2_b = (const float*)d_in[4];
    const float* q_U = (const float*)d_in[5];
    const float* q_V = (const float*)d_in[6];
    const float* q_b = (const float*)d_in[7];
    const float* k_U = (const float*)d_in[8];
    const float* k_V = (const float*)d_in[9];
    const float* k_b = (const float*)d_in[10];
    const float* v_U = (const float*)d_in[11];
    const float* v_V = (const float*)d_in[12];
    const float* v_b = (const float*)d_in[13];
    const float* out_U = (const float*)d_in[14];
    const float* out_V = (const float*)d_in[15];
    const float* out_b = (const float*)d_in[16];
    const float* fc1_U = (const float*)d_in[17];
    const float* fc1_V = (const float*)d_in[18];
    const float* fc1_b = (const float*)d_in[19];
    const float* fc2_U = (const float*)d_in[20];
    const float* fc2_V = (const float*)d_in[21];
    const float* fc2_b = (const float*)d_in[22];
    float* outp = (float*)d_out;

    char* ws = (char*)d_ws;
    size_t off = 0;
    auto alloc = [&](size_t bytes) -> char* {
        char* p = ws + off;
        off += (bytes + 255) & ~(size_t)255;
        return p;
    };
    // activations
    _Float16* x16  = (_Float16*)alloc((size_t)BS * D * 2);        // LN out (reused for LN2)
    _Float16* Ppad = (_Float16*)alloc((size_t)BS * 3 * D * 2);    // padded qkv latents (24MB)
    _Float16* Q16  = (_Float16*)alloc((size_t)BH * S * dh * 2);
    _Float16* K16  = (_Float16*)alloc((size_t)BH * S * dh * 2);
    _Float16* Vt16 = (_Float16*)alloc((size_t)BH * dh * S * 2);
    _Float16* ctx  = (_Float16*)alloc((size_t)BS * D * 2);
    _Float16* t1   = (_Float16*)alloc((size_t)BS * r_out * 2);
    float*    h    = (float*)   alloc((size_t)BS * D * 4);
    // packed f16 weights (transposed to [N][K])
    _Float16* qkvUt  = (_Float16*)alloc((size_t)3 * D * D * 2);   // [3*1024][1024]
    _Float16* headWt = (_Float16*)alloc((size_t)3 * H * 64 * 64 * 2);
    _Float16* outUt  = (_Float16*)alloc((size_t)r_out * D * 2);
    _Float16* outVt  = (_Float16*)alloc((size_t)D * r_out * 2);
    _Float16* fc1Ut  = (_Float16*)alloc((size_t)r_mlp * D * 2);
    _Float16* fc1Vt  = (_Float16*)alloc((size_t)I * r_mlp * 2);
    _Float16* fc2Ut  = (_Float16*)alloc((size_t)r_mlp * I * 2);
    _Float16* fc2Vt  = (_Float16*)alloc((size_t)D * r_mlp * 2);
    // region reuse (stream-ordered, producers dead before consumers)
    _Float16* x2_16 = x16;
    _Float16* tm1   = Ppad;                       // Ppad dead after head_proj
    _Float16* tm2   = Ppad + (size_t)BS * r_mlp;
    _Float16* hmid  = Q16;                        // Q/K/Vt/ctx = exactly BS*I*2

    // ---- weight packing ----
    auto nb = [](long n) { return (unsigned)((n + 255) / 256); };
    pack_U_pad<<<nb((long)D * D), 256, 0, stream>>>(q_U, qkvUt,             D, H, r);
    pack_U_pad<<<nb((long)D * D), 256, 0, stream>>>(k_U, qkvUt + (size_t)D * D,     D, H, r);
    pack_U_pad<<<nb((long)D * D), 256, 0, stream>>>(v_U, qkvUt + (size_t)2 * D * D, D, H, r);
    pack_headW_pad<<<nb((long)H * 64 * 64), 256, 0, stream>>>(q_V, headWt,                    H, r);
    pack_headW_pad<<<nb((long)H * 64 * 64), 256, 0, stream>>>(k_V, headWt + (size_t)H * 4096, H, r);
    pack_headW_pad<<<nb((long)H * 64 * 64), 256, 0, stream>>>(v_V, headWt + (size_t)2 * H * 4096, H, r);
    pack_transpose_f16<<<nb((long)D * r_out), 256, 0, stream>>>(out_U, outUt, D, r_out);
    pack_transpose_f16<<<nb((long)r_out * D), 256, 0, stream>>>(out_V, outVt, r_out, D);
    pack_transpose_f16<<<nb((long)D * r_mlp), 256, 0, stream>>>(fc1_U, fc1Ut, D, r_mlp);
    pack_transpose_f16<<<nb((long)r_mlp * I), 256, 0, stream>>>(fc1_V, fc1Vt, r_mlp, I);
    pack_transpose_f16<<<nb((long)I * r_mlp), 256, 0, stream>>>(fc2_U, fc2Ut, I, r_mlp);
    pack_transpose_f16<<<nb((long)r_mlp * D), 256, 0, stream>>>(fc2_V, fc2Vt, r_mlp, D);

    // ---- forward pass ----
    ln_to_f16<<<BS, 256, 0, stream>>>(hidden, ln1_g, ln1_b, x16, D);

    // fused qkv latent projection: [4096 x 3072] = x16 @ qkvUt^T, K=1024
    gemm_wmma<false,false,false,true><<<dim3(3 * D / 64, BS / 128), 128, 0, stream>>>(
        x16, qkvUt, nullptr, nullptr, Ppad, nullptr, BS, 3 * D, D);

    dim3 gHd(BS / 128, H);
    head_proj<false><<<gHd, 128, 0, stream>>>(Ppad, 3 * D, 0,     headWt,                    q_b, Q16,  S, H);
    head_proj<false><<<gHd, 128, 0, stream>>>(Ppad, 3 * D, D,     headWt + (size_t)H * 4096, k_b, K16,  S, H);
    head_proj<true ><<<gHd, 128, 0, stream>>>(Ppad, 3 * D, 2 * D, headWt + (size_t)2 * H * 4096, v_b, Vt16, S, H);

    flash_attn<<<dim3(S / 64, BH), 128, 0, stream>>>(Q16, K16, Vt16, ctx, S, H);

    gemm_wmma<false,false,false,true><<<dim3(r_out / 64, BS / 128), 128, 0, stream>>>(
        ctx, outUt, nullptr, nullptr, t1, nullptr, BS, r_out, D);
    gemm_wmma<true,false,true,false><<<dim3(D / 64, BS / 128), 128, 0, stream>>>(
        t1, outVt, out_b, hidden, nullptr, h, BS, D, r_out);

    ln_to_f16<<<BS, 256, 0, stream>>>(h, ln2_g, ln2_b, x2_16, D);

    gemm_wmma<false,false,false,true><<<dim3(r_mlp / 64, BS / 128), 128, 0, stream>>>(
        x2_16, fc1Ut, nullptr, nullptr, tm1, nullptr, BS, r_mlp, D);
    gemm_wmma<true,true,false,true><<<dim3(I / 64, BS / 128), 128, 0, stream>>>(
        tm1, fc1Vt, fc1_b, nullptr, hmid, nullptr, BS, I, r_mlp);
    gemm_wmma<false,false,false,true><<<dim3(r_mlp / 64, BS / 128), 128, 0, stream>>>(
        hmid, fc2Ut, nullptr, nullptr, tm2, nullptr, BS, r_mlp, I);
    gemm_wmma<true,false,true,false><<<dim3(D / 64, BS / 128), 128, 0, stream>>>(
        tm2, fc2Vt, fc2_b, h, nullptr, outp, BS, D, r_mlp);
}
